// ResLinear_13408887898905
// MI455X (gfx1250) — matrix operand order repaired and checked by reading, not compiled, not run on hardware
//
#include <hip/hip_runtime.h>

typedef __attribute__((ext_vector_type(16))) _Float16 v16h;
typedef __attribute__((ext_vector_type(8)))  float    v8f;
typedef __attribute__((ext_vector_type(4)))  float    v4f;

#define HID 20
#define WAVES 8
#define TPW 4                           // tiles (of 16 points) per wave
#define PTS_PER_WG (WAVES * TPW * 16)   // 512

__device__ __forceinline__ v8f wmma_f16_c0(v16h a, v16h b) {
  // C = inline 0: v_wmma_f32_16x16x32_f16 v[D], v[A], v[B], 0
  v8f c = {};
  return __builtin_amdgcn_wmma_f32_16x16x32_f16(
      /*neg_a=*/false, a, /*neg_b=*/false, b,
      /*c_mod=*/(short)0, c, /*reuse_a=*/false, /*reuse_b=*/false);
}

__device__ __forceinline__ void swish8(v8f& c) {
#pragma unroll
  for (int r = 0; r < 8; ++r) {
    float v = c[r];
    // v * sigmoid(v) = v * rcp(1 + e^-v); denominator > 1 -> rcp approx fine
    c[r] = v * __builtin_amdgcn_rcpf(1.0f + __expf(-v));
  }
}

// Convert a 16x32 result held as two f32 C-layout tiles (cols 0..15 / 16..19+pad)
// into the f16 A-operand layout for the next layer, via a per-wave LDS bounce.
// A layout: lane m (m<16): halves 0..7 = K0..7, halves 8..15 = K16..23
//           lane m+16    : halves 0..7 = K8..15, halves 8..15 = K24..31
// Scratch region [0,512) holds the operand; [512,1024) is a trash window so the
// tile-1 scatter is BRANCHLESS: lanes n16>=4 (k=20..31, offsets < 512) are
// redirected there with one v_cndmask instead of an exec-mask branch.
// Invariants in [0,512): K>=21 slots stay 0; K==20 slot (lane<16, half 12) stays 1.0
// so the bias row staged in B performs the +bias.
// NOTE: no hardware wait needed between scatter and gather: LDS ops from one
// wave are architecturally in-order; only a compile-time barrier is required.
__device__ __forceinline__ v16h c_to_a(v8f c0, v8f c1, _Float16* sc, int lane) {
  const int g = lane >> 4, n16 = lane & 15;
  _Float16* dst1 = sc + ((n16 < 4) ? 0 : 512);
  {
    const int k  = n16;                       // tile 0: K = 0..15
    const int dl = (k >> 3) & 1;
    const int hi = (k & 7) + ((k >> 4) << 3);
#pragma unroll
    for (int r = 0; r < 8; ++r)
      sc[((r + 8 * g) + 16 * dl) * 16 + hi] = (_Float16)c0[r];
  }
  {
    const int k  = 16 + n16;                  // tile 1: K = 16..31 (valid: 16..19)
    const int dl = (k >> 3) & 1;
    const int hi = (k & 7) + ((k >> 4) << 3);
#pragma unroll
    for (int r = 0; r < 8; ++r)
      dst1[((r + 8 * g) + 16 * dl) * 16 + hi] = (_Float16)c1[r];
  }
  asm volatile("" ::: "memory");              // compile-time ordering only
  return *(const v16h*)(sc + lane * 16);
}

__global__ __launch_bounds__(256) void resnet_pinn_kernel(
    const float* __restrict__ x,
    const float* __restrict__ front_W, const float* __restrict__ front_b,
    const float* __restrict__ back_W,  const float* __restrict__ back_b,
    const float* __restrict__ Wn1, const float* __restrict__ bn1,
    const float* __restrict__ Wl1, const float* __restrict__ bl1,
    const float* __restrict__ Wn2, const float* __restrict__ bn2,
    const float* __restrict__ Wl2, const float* __restrict__ bl2,
    const float* __restrict__ Wn3, const float* __restrict__ bn3,
    const float* __restrict__ Wl3, const float* __restrict__ bl3,
    float* __restrict__ out)
{
  // B-operand layout per 32x16 tile: lane = n + 16*h, half i<8 -> k = 8h+i, i>=8 -> k = 16+8h+(i-8).
  // Row k==HID(20) of each staged B tile carries the bias vector (A has 1.0 at K=20).
  __shared__ __align__(32) _Float16 wB[4][2][32][16];   // 4 matrices of the active block
  __shared__ __align__(32) _Float16 fB[2][32][16];      // front_W tiles (bias in row k=4)
  __shared__ __align__(32) _Float16 bkB[32][16];        // back_W tile   (bias in row k=20)
  __shared__ __align__(32) _Float16 scr[WAVES][1024];   // per-wave bounce + trash window

  const int tid  = threadIdx.x;
  const int lane = tid & 31;
  const int wave = tid >> 5;
  const int g    = lane >> 4;
  const int n16  = lane & 15;
  _Float16* sc   = &scr[wave][0];

  // ---- one-time staging of front/back weights (bias folded into extra K row) ----
  if (tid < 64) {
    int tile = tid >> 5, l = tid & 31;
    int nn = (l & 15) + tile * 16, kh = l >> 4;
#pragma unroll
    for (int i = 0; i < 16; ++i) {
      int k = (i < 8) ? (kh * 8 + i) : (16 + kh * 8 + (i - 8));
      float v = 0.0f;
      if (nn < HID) {
        if (k < 4)       v = front_W[k * HID + nn];
        else if (k == 4) v = front_b[nn];
      }
      fB[tile][l][i] = (_Float16)v;
    }
  } else if (tid < 96) {
    int l = tid - 64;
    int nn = l & 15, kh = l >> 4;
#pragma unroll
    for (int i = 0; i < 16; ++i) {
      int k = (i < 8) ? (kh * 8 + i) : (16 + kh * 8 + (i - 8));
      float v = 0.0f;
      if (nn < 4) {
        if (k < HID)       v = back_W[k * 4 + nn];
        else if (k == HID) v = back_b[nn];
      }
      bkB[l][i] = (_Float16)v;
    }
  }
  // scratch init: zeros everywhere except the K==20 "bias one" slot (lane<16, half 12)
#pragma unroll
  for (int i = 0; i < 16; ++i) {
    sc[lane * 16 + i]       = (lane < 16 && i == 12) ? (_Float16)1.0f : (_Float16)0.0f;
    sc[512 + lane * 16 + i] = (_Float16)0.0f;   // trash window
  }
  __syncthreads();

  const long p0base = (long)blockIdx.x * PTS_PER_WG + (long)wave * (TPW * 16);
  __builtin_prefetch(x + p0base * 4, 0, 0);   // global_prefetch_b8 of our x region

  v16h h0A[TPW], curA[TPW];
  v8f  acc0[TPW], acc1[TPW];

  // ---- front layer: h0 = x @ front_W + front_b (no activation), branchless ----
  {
    v16h fb0 = *(const v16h*)&fB[0][lane][0];
    v16h fb1 = *(const v16h*)&fB[1][lane][0];
    const bool lo = (lane < 16);
#pragma unroll
    for (int t = 0; t < TPW; ++t) {
      long p0 = p0base + t * 16;
      v4f xv = *(const v4f*)(x + (p0 + (lane & 15)) * 4);  // lanes 16..31 duplicate
      v16h ax;
#pragma unroll
      for (int i = 0; i < 16; ++i) ax[i] = (_Float16)0.0f;
      ax[0] = lo ? (_Float16)xv.x : (_Float16)0.0f;
      ax[1] = lo ? (_Float16)xv.y : (_Float16)0.0f;
      ax[2] = lo ? (_Float16)xv.z : (_Float16)0.0f;
      ax[3] = lo ? (_Float16)xv.w : (_Float16)0.0f;
      ax[4] = lo ? (_Float16)1.0f : (_Float16)0.0f;        // bias row selector (K=4)
      v8f c0 = wmma_f16_c0(ax, fb0);
      v8f c1 = wmma_f16_c0(ax, fb1);
      h0A[t] = c_to_a(c0, c1, sc, lane);
#pragma unroll
      for (int r = 0; r < 8; ++r) { acc0[t][r] = 0.0f; acc1[t][r] = 0.0f; }
    }
  }

  // ---- the three residual chains ----
  const float* Wns[3] = {Wn1, Wn2, Wn3};
  const float* bns[3] = {bn1, bn2, bn3};
  const float* Wls[3] = {Wl1, Wl2, Wl3};
  const float* bls[3] = {bl1, bl2, bl3};
  const int    lens[3] = {16, 4, 1};

  for (int c = 0; c < 3; ++c) {
#pragma unroll
    for (int t = 0; t < TPW; ++t) curA[t] = h0A[t];
    const int L = lens[c];
    for (int b = 0; b < L; ++b) {
      __syncthreads();   // previous block's compute done before restaging
      {
        // cooperative staging: 4 matrices x 2 tiles x 32 lanes == 256 threads
        int mi = tid >> 6, tile = (tid >> 5) & 1, l = tid & 31;
        int nn = tile * 16 + (l & 15), kh = l >> 4;
        const float* W = (mi < 3) ? (Wns[c] + ((long)b * 3 + mi) * (HID * HID))
                                  : (Wls[c] + (long)b * (HID * HID));
#pragma unroll
        for (int i = 0; i < 16; ++i) {
          int k = (i < 8) ? (kh * 8 + i) : (16 + kh * 8 + (i - 8));
          float v = 0.0f;
          if (nn < HID) {
            if (k < HID)       v = W[k * HID + nn];
            else if (k == HID) v = (mi < 3) ? bns[c][((long)b * 3 + mi) * HID + nn]
                                            : bls[c][(long)b * HID + nn];
          }
          wB[mi][tile][l][i] = (_Float16)v;
        }
      }
      __syncthreads();

      // hoist all 8 B operands of this block into VGPRs (shared by all 4 tiles)
      v16h Bm[4][2];
#pragma unroll
      for (int mi = 0; mi < 4; ++mi)
#pragma unroll
        for (int ti = 0; ti < 2; ++ti)
          Bm[mi][ti] = *(const v16h*)&wB[mi][ti][lane][0];

      const bool last = (b == L - 1);
#pragma unroll
      for (int t = 0; t < TPW; ++t) {
        v16h h = curA[t];
        v16h o = h;
        v8f cs0, cs1;
#pragma unroll
        for (int j = 0; j < 3; ++j) {
          v8f c0 = wmma_f16_c0(o, Bm[j][0]);
          v8f c1 = wmma_f16_c0(o, Bm[j][1]);
          swish8(c0); swish8(c1);
          if (j < 2) { o = c_to_a(c0, c1, sc, lane); }
          else       { cs0 = c0; cs1 = c1; }
        }
        // shortcut: swish(h @ Wl + bl)
        v8f s0 = wmma_f16_c0(h, Bm[3][0]);
        v8f s1 = wmma_f16_c0(h, Bm[3][1]);
        swish8(s0); swish8(s1);
#pragma unroll
        for (int r = 0; r < 8; ++r) { cs0[r] += s0[r]; cs1[r] += s1[r]; }
        if (last) {
#pragma unroll
          for (int r = 0; r < 8; ++r) { acc0[t][r] += cs0[r]; acc1[t][r] += cs1[r]; }
        } else {
          curA[t] = c_to_a(cs0, cs1, sc, lane);
        }
      }
    }
  }

  // ---- back layer: out = (x1+x2+x3) @ back_W + back_b ----
  v16h bk = *(const v16h*)&bkB[lane][0];
#pragma unroll
  for (int t = 0; t < TPW; ++t) {
    long p0 = p0base + t * 16;
    v16h aOut = c_to_a(acc0[t], acc1[t], sc, lane);   // includes 1.0 at K=20 -> +back_b
    v8f c = wmma_f16_c0(aOut, bk);
    if (n16 < 4) {
#pragma unroll
      for (int r = 0; r < 8; ++r)
        out[(p0 + r + 8 * g) * 4 + n16] = c[r];
    }
  }
}

extern "C" void kernel_launch(void* const* d_in, const int* in_sizes, int n_in,
                              void* d_out, int out_size, void* d_ws, size_t ws_size,
                              hipStream_t stream) {
  const float* x       = (const float*)d_in[0];
  const float* front_W = (const float*)d_in[1];
  const float* front_b = (const float*)d_in[2];
  const float* back_W  = (const float*)d_in[3];
  const float* back_b  = (const float*)d_in[4];
  const float* Wn1 = (const float*)d_in[5];
  const float* bn1 = (const float*)d_in[6];
  const float* Wl1 = (const float*)d_in[7];
  const float* bl1 = (const float*)d_in[8];
  const float* Wn2 = (const float*)d_in[9];
  const float* bn2 = (const float*)d_in[10];
  const float* Wl2 = (const float*)d_in[11];
  const float* bl2 = (const float*)d_in[12];
  const float* Wn3 = (const float*)d_in[13];
  const float* bn3 = (const float*)d_in[14];
  const float* Wl3 = (const float*)d_in[15];
  const float* bl3 = (const float*)d_in[16];
  float* out = (float*)d_out;

  const int n = in_sizes[0] / 4;            // 2,097,152 points
  const int grid = n / PTS_PER_WG;          // 4096 workgroups
  resnet_pinn_kernel<<<dim3(grid), dim3(256), 0, stream>>>(
      x, front_W, front_b, back_W, back_b,
      Wn1, bn1, Wl1, bl1,
      Wn2, bn2, Wl2, bl2,
      Wn3, bn3, Wl3, bl3,
      out);
}
